// GraphNetwork_30949534335567
// MI455X (gfx1250) — matrix operand order repaired and checked by reading, not compiled, not run on hardware
//
#include <hip/hip_runtime.h>
#include <hip/hip_bf16.h>

// ============================================================================
// GraphNetwork for MI455X (gfx1250, wave32, WMMA).
//  - GEMMs via v_wmma_f32_16x16x32_bf16 (bf16 operands, f32 accumulate).
//  - B operands pre-packed on device into ISA wave32 fragment order
//    -> each fragment is one aligned 32B load (2x global_load_b128).
//  - A fragments vector-loaded (b128) from point-major bf16 activations.
//  - Edge-conv1 generates |n_i - n_j| ++ boundary on the fly in the A-load.
//  - BatchNorm batch stats fused into each GEMM via f32 atomics from the
//    f32 WMMA accumulators; BN+LeakyReLU of conv k fused into A-load of k+1.
//  - bf16 ping-pong activation buffers (128MB each); workspace ~265 MB.
// ============================================================================

typedef __bf16 bf16;
typedef __attribute__((ext_vector_type(16))) __bf16 v16bf;
typedef __attribute__((ext_vector_type(8)))  __bf16 v8bf;
typedef __attribute__((ext_vector_type(8)))  float  v8f;
typedef __attribute__((ext_vector_type(4)))  float  v4f;

static constexpr int  NB   = 2;     // batch
static constexpr int  NN_  = 512;   // nodes
static constexpr int  NF_  = 128;   // node features
static constexpr long PTS  = (long)NB * NN_ * NN_;   // 524288 edge points
static constexpr float EPS_BN = 1e-5f;
static constexpr float EPS_LN = 1e-5f;
static constexpr float SLOPE  = 0.01f;

__device__ __forceinline__ float b2f(bf16 x)  { return (float)x; }
__device__ __forceinline__ bf16  f2b(float x) { return (bf16)x; }

__device__ __forceinline__ float block_reduce_add(float v, float* s) {
  int t = threadIdx.x;
  s[t] = v; __syncthreads();
  for (int k = 128; k > 0; k >>= 1) { if (t < k) s[t] += s[t + k]; __syncthreads(); }
  float r = s[0]; __syncthreads();
  return r;
}

// ---------------------------------------------------------------------------
// Generic bf16 WMMA GEMM:  Out[p, co] = sum_c A[p, c] * B[c, co]
//   MODE 0: A generated on the fly: c<128 -> |node[b,i,c]-node[b,j,c]|,
//           c in {128,129} -> boundary[b][c-128][i][j], else 0 (K padded).
//   MODE 1: A = bf16 buffer with fused BatchNorm (mean/rstd) + LeakyReLU.
//   MODE 2: A = plain bf16 buffer.
// B is fragment-packed: Bp[((kt*ntt + nt)*32 + lane)*16 + e], so each wave
// fetches a whole 32x16 B tile as one contiguous v16bf per lane.
// One wave = 16-row strip x NTILES 16-col tiles (A fragment reused).
// Optional per-output-channel sum/sumsq accumulation (BN stats).
// ---------------------------------------------------------------------------
template<int MODE, int NTILES>
__global__ __launch_bounds__(256)
void k_gemm(const bf16* __restrict__ A, int lda,
            const float* __restrict__ bn_mean, const float* __restrict__ bn_rstd,
            const bf16* __restrict__ Bp, int ntt, int ntile0, int K,
            bf16* __restrict__ Out, int out_stride, int out_col0,
            float* __restrict__ ssum, float* __restrict__ ssq,
            const float* __restrict__ node, const float* __restrict__ bnd)
{
  const int  lane = threadIdx.x & 31;
  const int  wv   = threadIdx.x >> 5;
  const int  ll   = lane & 15;
  const int  hh   = lane >> 4;
  const long p0   = ((long)blockIdx.x * 8 + wv) * 16;
  const long row  = p0 + ll;

  v8f acc[NTILES];
  #pragma unroll
  for (int t = 0; t < NTILES; ++t)
    #pragma unroll
    for (int r = 0; r < 8; ++r) acc[t][r] = 0.f;

  const float* nirow = nullptr; const float* njrow = nullptr;
  int bb = 0, ii = 0, jj = 0;
  if (MODE == 0) {
    const long nn2 = (long)NN_ * NN_;
    bb = (int)(row / nn2);
    long rem = row - (long)bb * nn2;
    ii = (int)(rem >> 9);
    jj = (int)(rem & (NN_ - 1));
    nirow = node + ((long)bb * NN_ + ii) * NF_;
    njrow = node + ((long)bb * NN_ + jj) * NF_;
  }

  const int KT = K >> 5;
  for (int kt = 0; kt < KT; ++kt) {
    const int cb = kt * 32 + hh * 8;     // ISA A layout: lane<16 -> K cb..cb+7,
    v16bf a;                             //               cb+16..cb+23; lane>=16 +8
    if (MODE == 0) {
      #pragma unroll
      for (int g = 0; g < 2; ++g) {
        const int c0 = cb + g * 16;
        if (c0 + 7 < NF_) {
          const v4f x0 = *(const v4f*)(nirow + c0);
          const v4f x1 = *(const v4f*)(nirow + c0 + 4);
          const v4f y0 = *(const v4f*)(njrow + c0);
          const v4f y1 = *(const v4f*)(njrow + c0 + 4);
          #pragma unroll
          for (int e = 0; e < 4; ++e) {
            a[g * 8 + e]     = f2b(fabsf(x0[e] - y0[e]));
            a[g * 8 + 4 + e] = f2b(fabsf(x1[e] - y1[e]));
          }
        } else {
          #pragma unroll
          for (int e = 0; e < 8; ++e) {
            const int c = c0 + e; float v = 0.f;
            if (c < NF_)          v = fabsf(nirow[c] - njrow[c]);
            else if (c < NF_ + 2) v = bnd[(((long)bb * 2 + (c - NF_)) * NN_ + ii) * NN_ + jj];
            a[g * 8 + e] = f2b(v);
          }
        }
      }
    } else {
      const bf16* ap = A + row * (long)lda + cb;
      if (kt + 1 < KT) __builtin_prefetch(ap + 32, 0, 1);   // global_prefetch_b8
      const v8bf lo = *(const v8bf*)ap;          // b128
      const v8bf hi = *(const v8bf*)(ap + 16);   // b128
      if (MODE == 2) {
        a = __builtin_shufflevector(lo, hi, 0, 1, 2, 3, 4, 5, 6, 7,
                                            8, 9, 10, 11, 12, 13, 14, 15);
      } else {
        #pragma unroll
        for (int e = 0; e < 8; ++e) {
          float v = b2f(lo[e]);
          int   c = cb + e;
          v = (v - bn_mean[c]) * bn_rstd[c];
          a[e] = f2b(v >= 0.f ? v : SLOPE * v);
          float w = b2f(hi[e]);
          c = cb + 16 + e;
          w = (w - bn_mean[c]) * bn_rstd[c];
          a[8 + e] = f2b(w >= 0.f ? w : SLOPE * w);
        }
      }
    }
    const bf16* wbase = Bp + (((long)kt * ntt + ntile0) * 32 + lane) * 16;
    #pragma unroll
    for (int nt = 0; nt < NTILES; ++nt) {
      const v16bf bfr = *(const v16bf*)(wbase + (long)nt * 32 * 16);  // 2x b128
      acc[nt] = __builtin_amdgcn_wmma_f32_16x16x32_bf16(
          false, a, false, bfr, (short)0, acc[nt], false, false);
    }
  }

  const int statcol0 = ntile0 * 16;
  #pragma unroll
  for (int nt = 0; nt < NTILES; ++nt) {
    const int col = nt * 16 + ll;
    float s = 0.f, s2 = 0.f;
    #pragma unroll
    for (int r = 0; r < 8; ++r) {        // ISA D layout: elem r -> row r+hh*8, col ll
      const float v = acc[nt][r];
      const long  m = p0 + r + hh * 8;
      Out[m * (long)out_stride + out_col0 + col] = f2b(v);
      s += v; s2 += v * v;
    }
    if (ssum) {
      atomicAdd(&ssum[statcol0 + col], s);
      atomicAdd(&ssq [statcol0 + col], s2);
    }
  }
}

// ---- pack conv weight [Co,Ci] f32 (transposed) into WMMA B-fragment order --
__global__ void k_wpack(const float* __restrict__ W, bf16* __restrict__ Wp,
                        int Co, int Ci, int KT)
{
  const int NT  = Co >> 4;
  const int idx = blockIdx.x * 256 + threadIdx.x;   // (kt*NT + nt)*32 + lane
  if (idx >= KT * NT * 32) return;
  const int lane = idx & 31;
  const int tile = idx >> 5;
  const int nt   = tile % NT;
  const int kt   = tile / NT;
  const int col  = nt * 16 + (lane & 15);
  const int kb   = kt * 32 + (lane >> 4) * 8;
  bf16* dst = Wp + (long)idx * 16;
  #pragma unroll
  for (int e = 0; e < 16; ++e) {
    const int k = kb + (e < 8 ? e : e + 8);
    dst[e] = (k < Ci) ? f2b(W[col * Ci + k]) : f2b(0.f);
  }
}

// ---- pack bf16 row-major [K,Co] matrix into WMMA B-fragment order ----------
__global__ void k_bpack(const bf16* __restrict__ S, bf16* __restrict__ Bp,
                        int Co, int KT)
{
  const int NT  = Co >> 4;
  const int idx = blockIdx.x * 256 + threadIdx.x;
  if (idx >= KT * NT * 32) return;
  const int lane = idx & 31;
  const int tile = idx >> 5;
  const int nt   = tile % NT;
  const int kt   = tile / NT;
  const int col  = nt * 16 + (lane & 15);
  const int kb   = kt * 32 + (lane >> 4) * 8;
  bf16* dst = Bp + (long)idx * 16;
  #pragma unroll
  for (int e = 0; e < 16; ++e) dst[e] = S[(long)(kb + (e < 8 ? e : e + 8)) * Co + col];
}

// ---- LayerNorm over (N,F) jointly per batch --------------------------------
__global__ __launch_bounds__(256)
void k_layernorm(const float* __restrict__ in, float* __restrict__ outf,
                 bf16* __restrict__ outb)
{
  __shared__ float sred[256];
  const int b = blockIdx.x;
  const int n = NN_ * NF_;
  const float* x = in + (long)b * n;
  float s = 0.f, s2 = 0.f;
  for (int i = threadIdx.x; i < n; i += 256) { float v = x[i]; s += v; s2 += v * v; }
  s  = block_reduce_add(s,  sred);
  s2 = block_reduce_add(s2, sred);
  const float mean = s / n;
  const float rstd = rsqrtf(s2 / n - mean * mean + EPS_LN);
  for (int i = threadIdx.x; i < n; i += 256) {
    const float v = (x[i] - mean) * rstd;
    outf[(long)b * n + i] = v;
    outb[(long)b * n + i] = f2b(v);
  }
}

// ---- L1-normalize (diag-masked) edge rows -> bf16 --------------------------
__global__ __launch_bounds__(256)
void k_l1norm(const float* __restrict__ ef, bf16* __restrict__ out)
{
  __shared__ float sred[256];
  const long base = (long)blockIdx.x * NN_;
  const int  i    = blockIdx.x & (NN_ - 1);
  float v0, v1, s = 0.f;
  { const int j = threadIdx.x;       v0 = (j == i) ? 0.f : ef[base + j]; s += fabsf(v0); }
  { const int j = threadIdx.x + 256; v1 = (j == i) ? 0.f : ef[base + j]; s += fabsf(v1); }
  s = block_reduce_add(s, sred);
  const float inv = 1.f / fmaxf(s, 1e-12f);
  out[base + threadIdx.x]       = f2b(v0 * inv);
  out[base + threadIdx.x + 256] = f2b(v1 * inv);
}

// ---- copy LayerNorm'd node features into cols 0..127 of node-MLP input -----
__global__ void k_copy_ln(const bf16* __restrict__ ln, bf16* __restrict__ X)
{
  const long idx = (long)blockIdx.x * 256 + threadIdx.x;   // B*N*128
  X[(idx >> 7) * 384 + (int)(idx & 127)] = ln[idx];
}

// ---- BN stats finalize -----------------------------------------------------
__global__ void k_finalize(const float* __restrict__ ssum, const float* __restrict__ ssq,
                           float* __restrict__ mean, float* __restrict__ rstd,
                           int C, float inv_count)
{
  const int c = threadIdx.x;
  if (c < C) {
    const float m = ssum[c] * inv_count;
    mean[c] = m;
    rstd[c] = rsqrtf(ssq[c] * inv_count - m * m + EPS_BN);
  }
}

// ---- BN+LeakyReLU of final node conv -> f32 (ws copy + d_out copy) ---------
__global__ void k_bn_node(const bf16* __restrict__ Y, const float* __restrict__ mean,
                          const float* __restrict__ rstd,
                          float* __restrict__ ws_out, float* __restrict__ out_node)
{
  const long idx = (long)blockIdx.x * 256 + threadIdx.x;   // B*N*128
  const int  c   = (int)(idx & 127);
  float v = (b2f(Y[idx]) - mean[c]) * rstd[c];
  v = v >= 0.f ? v : SLOPE * v;
  ws_out[idx] = v;
  out_node[idx] = v;
}

// ---- sigmoid(conv1x1(bn+lrelu(x), w_out) + b) ------------------------------
__global__ __launch_bounds__(256)
void k_sim(const bf16* __restrict__ Y, const float* __restrict__ mean,
           const float* __restrict__ rstd, const float* __restrict__ w,
           const float* __restrict__ bias, float* __restrict__ sim)
{
  const long p = (long)blockIdx.x * 256 + threadIdx.x;
  const bf16* y = Y + p * 64;
  float s = 0.f;
  #pragma unroll
  for (int c = 0; c < 64; ++c) {
    float v = (b2f(y[c]) - mean[c]) * rstd[c];
    v = v >= 0.f ? v : SLOPE * v;
    s += v * w[c];
  }
  s += bias[0];
  sim[p] = 1.f / (1.f + expf(-s));
}

// ---- edge merge: gate, L1-renorm, +force+eps, channel-normalize, *adj ------
__global__ __launch_bounds__(256)
void k_edge_merge(const float* __restrict__ ef, const float* __restrict__ sim,
                  const float* __restrict__ adj, float* __restrict__ out)
{
  __shared__ float sred[256];
  const int  b = blockIdx.x >> 9;
  const int  i = blockIdx.x & (NN_ - 1);
  const long base0 = (((long)b * 2 + 0) * NN_ + i) * NN_;
  const long base1 = (((long)b * 2 + 1) * NN_ + i) * NN_;
  const long sbase = ((long)b * NN_ + i) * NN_;
  float e0[2], e1[2], n0[2], n1[2];
  float ms0 = 0.f, ms1 = 0.f, d0 = 0.f, d1 = 0.f;
  #pragma unroll
  for (int t = 0; t < 2; ++t) {
    const int j = threadIdx.x + t * 256;
    const float m = (j == i) ? 0.f : 1.f;
    e0[t] = ef[base0 + j] * m;
    e1[t] = ef[base1 + j] * m;
    const float sj = sim[sbase + j];
    n0[t] = sj * e0[t];
    n1[t] = (1.f - sj) * e1[t];
    ms0 += e0[t]; ms1 += e1[t];
    d0 += fabsf(n0[t]); d1 += fabsf(n1[t]);
  }
  ms0 = block_reduce_add(ms0, sred);
  ms1 = block_reduce_add(ms1, sred);
  d0  = fmaxf(block_reduce_add(d0, sred), 1e-12f);
  d1  = fmaxf(block_reduce_add(d1, sred), 1e-12f);
  #pragma unroll
  for (int t = 0; t < 2; ++t) {
    const int j = threadIdx.x + t * 256;
    const float r0 = n0[t] / d0 * ms0 + ((j == i) ? 1.f : 0.f) + 1e-6f;
    const float r1 = n1[t] / d1 * ms1 + 1e-6f;
    const float inv = 1.f / (r0 + r1);
    out[base0 + j] = r0 * inv * adj[base0 + j];
    out[base1 + j] = r1 * inv * adj[base1 + j];
  }
}

// ============================================================================
extern "C" void kernel_launch(void* const* d_in, const int* in_sizes, int n_in,
                              void* d_out, int out_size, void* d_ws, size_t ws_size,
                              hipStream_t stream)
{
  (void)in_sizes; (void)n_in; (void)out_size; (void)ws_size;
  // Input order: node_feat, edge_feat, adj, adj_boundary, then per layer
  // (pytree sorted-key order): edge_b_out, edge_convs[0..3], edge_w_out,
  // node_convs[0..1].
  const float* node_in = (const float*)d_in[0];
  const float* edge_in = (const float*)d_in[1];
  const float* adj     = (const float*)d_in[2];
  const float* bndry   = (const float*)d_in[3];
  float* out       = (float*)d_out;
  float* out_edges = out;                                    // [L,B,2,N,N]
  float* out_nodes = out + (long)2 * NB * 2 * NN_ * NN_;     // [L,B,N,128]

  char* ws = (char*)d_ws;
  size_t off = 0;
  auto alloc = [&](size_t bytes) -> void* {
    off = (off + 255) & ~(size_t)255;
    void* p = ws + off; off += bytes; return p;
  };
  auto fragN = [](int KT, int NT) -> size_t { return (size_t)KT * NT * 32 * 16; };

  float* ln_f    = (float*)alloc((size_t)NB * NN_ * NF_ * 4);
  bf16*  ln_b    = (bf16*) alloc((size_t)NB * NN_ * NF_ * 2);
  float* nodeout = (float*)alloc((size_t)NB * NN_ * NF_ * 4);
  bf16*  efn     = (bf16*) alloc((size_t)NB * 2 * NN_ * NN_ * 2);
  bf16*  Xn      = (bf16*) alloc((size_t)NB * NN_ * 384 * 2);
  bf16*  Yn1     = (bf16*) alloc((size_t)NB * NN_ * 256 * 2);
  bf16*  Yn2     = (bf16*) alloc((size_t)NB * NN_ * 128 * 2);
  float* simb    = (float*)alloc((size_t)PTS * 4);
  bf16*  E0      = (bf16*) alloc((size_t)PTS * 128 * 2);     // 128 MB
  bf16*  E1      = (bf16*) alloc((size_t)PTS * 128 * 2);     // 128 MB
  bf16*  Bln[NB];                                            // packed ln (aggr B)
  for (int b = 0; b < NB; ++b) Bln[b] = (bf16*)alloc(fragN(16, 8) * 2);
  bf16 *We1[2], *We2[2], *We3[2], *We4[2], *Wn1[2], *Wn2[2];
  for (int l = 0; l < 2; ++l) {
    We1[l] = (bf16*)alloc(fragN(5, 8)  * 2);   // 160x128
    We2[l] = (bf16*)alloc(fragN(4, 8)  * 2);   // 128x128
    We3[l] = (bf16*)alloc(fragN(4, 4)  * 2);   // 128x64
    We4[l] = (bf16*)alloc(fragN(2, 4)  * 2);   // 64x64
    Wn1[l] = (bf16*)alloc(fragN(12, 16) * 2);  // 384x256
    Wn2[l] = (bf16*)alloc(fragN(8, 8)  * 2);   // 256x128
  }
  float* st_sum  = (float*)alloc(6 * 256 * 4);
  float* st_sq   = (float*)alloc(6 * 256 * 4);
  float* st_mean = (float*)alloc(6 * 256 * 4);
  float* st_rstd = (float*)alloc(6 * 256 * 4);

  // ---- device-side weight transpose + fragment pack (both layers) ----
  auto wpack = [&](const float* W, bf16* Wp, int Co, int Ci, int KT) {
    const int n = KT * (Co >> 4) * 32;
    k_wpack<<<(n + 255) / 256, 256, 0, stream>>>(W, Wp, Co, Ci, KT);
  };
  for (int l = 0; l < 2; ++l) {
    wpack((const float*)d_in[4 + l * 8 + 1], We1[l], 128, 130, 5);
    wpack((const float*)d_in[4 + l * 8 + 2], We2[l], 128, 128, 4);
    wpack((const float*)d_in[4 + l * 8 + 3], We3[l],  64, 128, 4);
    wpack((const float*)d_in[4 + l * 8 + 4], We4[l],  64,  64, 2);
    wpack((const float*)d_in[4 + l * 8 + 6], Wn1[l], 256, 384, 12);
    wpack((const float*)d_in[4 + l * 8 + 7], Wn2[l], 128, 256, 8);
  }

  const float* cur_node = node_in;
  const float* cur_edge = edge_in;
  for (int l = 0; l < 2; ++l) {
    const float* p_eb   = (const float*)d_in[4 + l * 8 + 0];
    const float* p_wout = (const float*)d_in[4 + l * 8 + 5];
    float* oel = out_edges + (long)l * NB * 2 * NN_ * NN_;
    float* onl = out_nodes + (long)l * NB * NN_ * NF_;

    hipMemsetAsync(st_sum, 0, 6 * 256 * 4, stream);
    hipMemsetAsync(st_sq,  0, 6 * 256 * 4, stream);

    // ------------------- node path -------------------
    k_layernorm<<<NB, 256, 0, stream>>>(cur_node, ln_f, ln_b);
    for (int b = 0; b < NB; ++b)       // pack ln as WMMA B operand (K=512,Co=128)
      k_bpack<<<(16 * 8 * 32 + 255) / 256, 256, 0, stream>>>(
          ln_b + (long)b * NN_ * NF_, Bln[b], NF_, 16);
    k_l1norm<<<NB * 2 * NN_, 256, 0, stream>>>(cur_edge, efn);
    for (int b = 0; b < NB; ++b)       // aggr = efn @ node_ln  -> Xn cols 128..383
      for (int k = 0; k < 2; ++k)
        k_gemm<2, 8><<<NN_ / 128, 256, 0, stream>>>(
            efn + (long)(b * 2 + k) * NN_ * NN_, NN_, nullptr, nullptr,
            Bln[b], 8, 0, NN_,
            Xn + (long)b * NN_ * 384, 384, 128 + k * 128,
            nullptr, nullptr, nullptr, nullptr);
    k_copy_ln<<<(NB * NN_ * NF_) / 256, 256, 0, stream>>>(ln_b, Xn);
    // node conv1: [1024,384] x [384,256], split into two 128-col halves
    k_gemm<2, 8><<<(NB * NN_) / 128, 256, 0, stream>>>(Xn, 384, nullptr, nullptr,
        Wn1[l], 16, 0, 384, Yn1, 256, 0, st_sum + 0 * 256, st_sq + 0 * 256,
        nullptr, nullptr);
    k_gemm<2, 8><<<(NB * NN_) / 128, 256, 0, stream>>>(Xn, 384, nullptr, nullptr,
        Wn1[l], 16, 8, 384, Yn1, 256, 128, st_sum + 0 * 256, st_sq + 0 * 256,
        nullptr, nullptr);
    k_finalize<<<1, 256, 0, stream>>>(st_sum + 0 * 256, st_sq + 0 * 256,
        st_mean + 0 * 256, st_rstd + 0 * 256, 256, 1.f / (NB * NN_));
    // node conv2: [1024,256] x [256,128] with fused BN+lrelu on input
    k_gemm<1, 8><<<(NB * NN_) / 128, 256, 0, stream>>>(Yn1, 256,
        st_mean + 0 * 256, st_rstd + 0 * 256,
        Wn2[l], 8, 0, 256, Yn2, 128, 0, st_sum + 1 * 256, st_sq + 1 * 256,
        nullptr, nullptr);
    k_finalize<<<1, 256, 0, stream>>>(st_sum + 1 * 256, st_sq + 1 * 256,
        st_mean + 1 * 256, st_rstd + 1 * 256, 128, 1.f / (NB * NN_));
    k_bn_node<<<(NB * NN_ * NF_) / 256, 256, 0, stream>>>(Yn2,
        st_mean + 1 * 256, st_rstd + 1 * 256, nodeout, onl);

    // ------------------- edge path -------------------
    // conv1: on-the-fly |n_i-n_j|++boundary (K padded 130->160) -> E0[.,128]
    k_gemm<0, 8><<<(int)(PTS / 128), 256, 0, stream>>>(nullptr, 0, nullptr, nullptr,
        We1[l], 8, 0, 160, E0, 128, 0, st_sum + 2 * 256, st_sq + 2 * 256,
        nodeout, bndry);
    k_finalize<<<1, 256, 0, stream>>>(st_sum + 2 * 256, st_sq + 2 * 256,
        st_mean + 2 * 256, st_rstd + 2 * 256, 128, 1.f / (float)PTS);
    k_gemm<1, 8><<<(int)(PTS / 128), 256, 0, stream>>>(E0, 128,
        st_mean + 2 * 256, st_rstd + 2 * 256,
        We2[l], 8, 0, 128, E1, 128, 0, st_sum + 3 * 256, st_sq + 3 * 256,
        nullptr, nullptr);
    k_finalize<<<1, 256, 0, stream>>>(st_sum + 3 * 256, st_sq + 3 * 256,
        st_mean + 3 * 256, st_rstd + 3 * 256, 128, 1.f / (float)PTS);
    k_gemm<1, 4><<<(int)(PTS / 128), 256, 0, stream>>>(E1, 128,
        st_mean + 3 * 256, st_rstd + 3 * 256,
        We3[l], 4, 0, 128, E0, 64, 0, st_sum + 4 * 256, st_sq + 4 * 256,
        nullptr, nullptr);
    k_finalize<<<1, 256, 0, stream>>>(st_sum + 4 * 256, st_sq + 4 * 256,
        st_mean + 4 * 256, st_rstd + 4 * 256, 64, 1.f / (float)PTS);
    k_gemm<1, 4><<<(int)(PTS / 128), 256, 0, stream>>>(E0, 64,
        st_mean + 4 * 256, st_rstd + 4 * 256,
        We4[l], 4, 0, 64, E1, 64, 0, st_sum + 5 * 256, st_sq + 5 * 256,
        nullptr, nullptr);
    k_finalize<<<1, 256, 0, stream>>>(st_sum + 5 * 256, st_sq + 5 * 256,
        st_mean + 5 * 256, st_rstd + 5 * 256, 64, 1.f / (float)PTS);
    k_sim<<<(int)(PTS / 256), 256, 0, stream>>>(E1,
        st_mean + 5 * 256, st_rstd + 5 * 256, p_wout, p_eb, simb);
    k_edge_merge<<<NB * NN_, 256, 0, stream>>>(cur_edge, simb, adj, oel);

    cur_node = nodeout;
    cur_edge = oel;      // next layer reads this layer's freshly written output
  }
}
